// KANLinear_29987461660768
// MI455X (gfx1250) — matrix operand order repaired and checked by reading, not compiled, not run on hardware
//
#include <hip/hip_runtime.h>
#include <hip/hip_bf16.h>

// ---------------------------------------------------------------------------
// KAN linear layer for MI455X (gfx1250, wave32, WMMA).
//   out[16384,512] = [silu(LN(x)) | bsplines(LN(x))] (16384x3584 bf16)
//                    @ [base_w | spline_w]^T (3584x512 bf16)  + bias   (f32 acc)
// Matrix-core bound (~60 GFLOP); all FLOPs through v_wmma_f32_16x16x32_bf16.
// GEMM: 64x64 wave tile (16 wmma / K-step), 128x256 block tile -> A streamed
// from HBM ~once, B panel L2-resident, ~512 B of fragments per WMMA.
// ---------------------------------------------------------------------------

typedef __attribute__((ext_vector_type(16))) __bf16 v16bf;
typedef __attribute__((ext_vector_type(8)))  float  v8f;

#define TOKENS      16384
#define FEAT        512
#define UNITS       512
#define NCOEF       6
#define KCOLS       3584            // 512 base + 512*6 spline
#define KTILES      112             // KCOLS / 32
#define NTILES      32              // UNITS / 16
#define APACK_BYTES ((size_t)(TOKENS/16) * KTILES * 32 * 16 * 2)   // 117,440,512
#define BPACK_BYTES ((size_t)NTILES * KTILES * 32 * 16 * 2)        //   3,670,016

// --- A-fragment swizzle (ISA 7.12.2, 16-bit A 16x32) -----------------------
// lane<16 : row=lane,    e0..7 -> K 0..7,  e8..15 -> K 16..23
// lane>=16: row=lane-16, e0..7 -> K 8..15, e8..15 -> K 24..31
__device__ __forceinline__ size_t apack_idx(int mt, int r16, int k) {
    int kt = k >> 5, kk = k & 31;
    int hi = (kk & 15) >> 3;                 // upper half of each 16-group
    int ln = r16 + (hi << 4);
    int e  = (kk & 7) | ((kk >> 4) << 3);
    return ((((size_t)mt * KTILES + kt) * 32) + ln) * 16 + e;
}

// ============================================================
// Kernel 0: pack weights fp32 -> bf16 in WMMA B-fragment layout
//   B tile (K=32 x N=16): lane = 16*((K/8)%2) + N%16, element = 8*(K/16) + K%8
// ============================================================
__global__ __launch_bounds__(256) void kan_pack_w(
    const float* __restrict__ base_w,     // [512, 512]
    const float* __restrict__ spline_w,   // [512, 3072]
    __bf16* __restrict__ Bpack)
{
    size_t idx = (size_t)blockIdx.x * 256 + threadIdx.x;   // NTILES*KTILES*32*16 total
    int e    = idx & 15;
    int lane = (idx >> 4) & 31;
    int rem  = (int)(idx >> 9);
    int kt   = rem % KTILES;
    int nt   = rem / KTILES;
    int u    = nt * 16 + (lane & 15);
    int k    = kt * 32 + ((e >> 3) << 4) + ((lane >> 4) << 3) + (e & 7);
    float v  = (k < FEAT) ? base_w[(size_t)u * FEAT + k]
                          : spline_w[(size_t)u * (FEAT * NCOEF) + (k - FEAT)];
    Bpack[idx] = (__bf16)v;
}

// ============================================================
// Kernel 1: LayerNorm + silu + cubic B-spline bases -> bf16 A panel
//   1 block = 8 waves = 16 tokens; each wave owns 2 tokens (shuffle LN).
//   Grid rows are a broadcast of one knot vector -> load once per thread,
//   precompute all Cox-de Boor reciprocal denominators (no divides in loop).
// ============================================================
__global__ __launch_bounds__(256) void kan_act(
    const float* __restrict__ x,        // [16384, 512]
    const float* __restrict__ grid,     // [512, 10] (identical rows)
    const float* __restrict__ gamma,    // [512]
    const float* __restrict__ beta,     // [512]
    __bf16* __restrict__ Apack)
{
    int lane = threadIdx.x & 31;
    int wave = threadIdx.x >> 5;
    int mt   = blockIdx.x;                       // 16-token tile

    // knot vector (all rows identical by construction)
    float g[10];
    #pragma unroll
    for (int t = 0; t < 10; ++t) g[t] = grid[t];

    // precompute recursion constants: 21 terms (k=1:8, k=2:7, k=3:6)
    //   left  = (xn - g[t])     / (g[t+k]   - g[t])   = xn*invL - cL
    //   right = (g[t+k+1] - xn) / (g[t+k+1] - g[t+1]) = cR - xn*invR
    float invL[21], cL[21], invR[21], cR[21];
    {
        int idx = 0;
        #pragma unroll
        for (int k = 1; k <= 3; ++k)
            #pragma unroll
            for (int t = 0; t + k < 9; ++t, ++idx) {
                invL[idx] = 1.0f / (g[t + k] - g[t]);
                cL[idx]   = g[t] * invL[idx];
                invR[idx] = 1.0f / (g[t + k + 1] - g[t + 1]);
                cR[idx]   = g[t + k + 1] * invR[idx];
            }
    }

    for (int half = 0; half < 2; ++half) {
        int r16 = wave * 2 + half;
        int tok = mt * 16 + r16;
        const float* xr = x + (size_t)tok * FEAT;

        float xs[16];
        float s = 0.f, s2 = 0.f;
        #pragma unroll
        for (int i = 0; i < 16; ++i) {
            float v = xr[lane + (i << 5)];
            xs[i] = v; s += v; s2 += v * v;
        }
        #pragma unroll
        for (int m = 16; m >= 1; m >>= 1) {
            s  += __shfl_xor(s,  m, 32);
            s2 += __shfl_xor(s2, m, 32);
        }
        float mean = s * (1.0f / FEAT);
        float var  = s2 * (1.0f / FEAT) - mean * mean;
        float rs   = rsqrtf(var + 1e-3f);

        #pragma unroll 4
        for (int i = 0; i < 16; ++i) {
            int f = lane + (i << 5);
            float xn = (xs[i] - mean) * rs * gamma[f] + beta[f];

            // base path: silu
            float si = xn / (1.0f + __expf(-xn));
            Apack[apack_idx(mt, r16, f)] = (__bf16)si;

            // spline path: Cox-de Boor (order 3), constants hoisted
            float b[9];
            #pragma unroll
            for (int t = 0; t < 9; ++t)
                b[t] = (xn >= g[t] && xn < g[t + 1]) ? 1.0f : 0.0f;
            {
                int idx = 0;
                #pragma unroll
                for (int k = 1; k <= 3; ++k)
                    #pragma unroll
                    for (int t = 0; t + k < 9; ++t, ++idx) {
                        float left  = fmaf(xn,  invL[idx], -cL[idx]);
                        float right = fmaf(-xn, invR[idx],  cR[idx]);
                        b[t] = left * b[t] + right * b[t + 1];
                    }
            }
            int kbase = FEAT + f * NCOEF;
            #pragma unroll
            for (int j = 0; j < NCOEF; ++j)
                Apack[apack_idx(mt, r16, kbase + j)] = (__bf16)b[j];
        }
    }
}

// ============================================================
// Kernel 2: bf16 WMMA GEMM  C[16384,512] = A[16384,3584] x B[3584,512] + bias
//   block: 8 waves (2 wm x 4 wn), tile 128M x 256N; wave: 64M x 64N (4x4)
//   grid: (2 N-blocks, 128 M-blocks)
// ============================================================
__global__ __launch_bounds__(256) void kan_gemm(
    const __bf16* __restrict__ Apack,
    const __bf16* __restrict__ Bpack,
    const float*  __restrict__ bias,    // [512]
    float* __restrict__ out)            // [16384, 512]
{
    int lane = threadIdx.x & 31;
    int wave = threadIdx.x >> 5;
    int wm = wave & 1, wn = wave >> 1;
    int mt0 = blockIdx.y * 8  + wm * 4;   // 4 M tiles of 16
    int nt0 = blockIdx.x * 16 + wn * 4;   // 4 N tiles of 16

    const v16bf* Ap = (const v16bf*)Apack;   // index: (mt*KTILES + kt)*32 + lane
    const v16bf* Bp = (const v16bf*)Bpack;   // index: (nt*KTILES + kt)*32 + lane

    v8f c[4][4] = {};

    for (int kt = 0; kt < KTILES; ++kt) {
        v16bf a[4], b[4];
        #pragma unroll
        for (int i = 0; i < 4; ++i)
            a[i] = Ap[(((size_t)(mt0 + i) * KTILES + kt) << 5) + lane];
        #pragma unroll
        for (int j = 0; j < 4; ++j)
            b[j] = Bp[(((size_t)(nt0 + j) * KTILES + kt) << 5) + lane];
        #pragma unroll
        for (int i = 0; i < 4; ++i)
            #pragma unroll
            for (int j = 0; j < 4; ++j)
                c[i][j] = __builtin_amdgcn_wmma_f32_16x16x32_bf16(
                    false, a[i], false, b[j], (short)0, c[i][j], false, false);
    }

    // epilogue: C/D layout -> row = v + 8*(lane/16), col = lane%16
    int ncol = lane & 15;
    int rhi  = lane >> 4;
    #pragma unroll
    for (int j = 0; j < 4; ++j) {
        int u = (nt0 + j) * 16 + ncol;
        float bv = bias[u];
        #pragma unroll
        for (int i = 0; i < 4; ++i) {
            int tokbase = (mt0 + i) * 16 + rhi * 8;
            #pragma unroll
            for (int v = 0; v < 8; ++v)
                out[(size_t)(tokbase + v) * UNITS + u] = c[i][j][v] + bv;
        }
    }
}

extern "C" void kernel_launch(void* const* d_in, const int* in_sizes, int n_in,
                              void* d_out, int out_size, void* d_ws, size_t ws_size,
                              hipStream_t stream) {
    const float* x      = (const float*)d_in[0];   // [8,2048,512]
    const float* grid   = (const float*)d_in[1];   // [512,10]
    const float* base_w = (const float*)d_in[2];   // [512,512]
    const float* bias   = (const float*)d_in[3];   // [512]
    const float* spl_w  = (const float*)d_in[4];   // [512,3072]
    const float* gamma  = (const float*)d_in[5];   // [512]
    const float* beta   = (const float*)d_in[6];   // [512]

    __bf16* Apack = (__bf16*)d_ws;
    __bf16* Bpack = (__bf16*)((char*)d_ws + APACK_BYTES);

    // pack weights: NTILES*KTILES*32*16 = 1,835,008 elements
    kan_pack_w<<<7168, 256, 0, stream>>>(base_w, spl_w, Bpack);
    // activations: 16384 tokens / 16 per block
    kan_act<<<1024, 256, 0, stream>>>(x, grid, gamma, beta, Apack);
    // GEMM: 2 N-blocks x 128 M-blocks
    dim3 g(2, 128);
    kan_gemm<<<g, 256, 0, stream>>>(Apack, Bpack, bias, (float*)d_out);
}
